// GraphSAGE_120259084718
// MI455X (gfx1250) — matrix operand order repaired and checked by reading, not compiled, not run on hardware
//
#include <hip/hip_runtime.h>
#include <hip/hip_fp16.h>

typedef __attribute__((ext_vector_type(16))) _Float16 v16h;
typedef __attribute__((ext_vector_type(8)))  _Float16 v8h;
typedef __attribute__((ext_vector_type(4)))  _Float16 v4h;
typedef __attribute__((ext_vector_type(8)))  float    v8f;

#define N_NODES 50000
#define N_EDGES 800000
#define IN_C    128
#define HID_C   128
#define OUT_C   64

// no-return fp32 global atomic add -> global_atomic_add_f32 (STOREcnt path)
static __device__ __forceinline__ void atomAddF(float* p, float v) {
    __hip_atomic_fetch_add(p, v, __ATOMIC_RELAXED, __HIP_MEMORY_SCOPE_AGENT);
}

// ---------------------------------------------------------------- zero scratch
__global__ void zero4_kernel(float4* __restrict__ p, long n4) {
    long i = (long)blockIdx.x * blockDim.x + threadIdx.x;
    long stride = (long)gridDim.x * blockDim.x;
    float4 z = {0.f, 0.f, 0.f, 0.f};
    for (; i < n4; i += stride) p[i] = z;
}

// ------------------------------------------------- pack weights into B-fragment
// dst layout: [colTile = O/16][kTile = K/32][lane 0..31][i 0..15] halves
// value at (ct,kt,lane,i) = W[o*K + k], o = ct*16 + (lane&15),
//                           k = 32*kt + (i>>3)*16 + (lane>>4)*8 + (i&7)
// -> per-lane B fragment for v_wmma_f32_16x16x32_f16 is 16 contiguous halves.
__global__ void pack_weights_kernel(const float* __restrict__ W,
                                    _Float16* __restrict__ dst, int O, int K) {
    int total = O * K;
    int ktiles = K >> 5;
    for (int t = blockIdx.x * blockDim.x + threadIdx.x; t < total;
         t += gridDim.x * blockDim.x) {
        int i    = t & 15;
        int lane = (t >> 4) & 31;
        int rest = t >> 9;
        int kt   = rest % ktiles;
        int ct   = rest / ktiles;
        int k = 32 * kt + ((i >> 3) << 4) + ((lane >> 4) << 3) + (i & 7);
        int o = ct * 16 + (lane & 15);
        dst[t] = (_Float16)W[o * K + k];
    }
}

// ------------------------------------------------------ edge scatter (layer 1)
// one wave per edge: 32 lanes x float4 = 128 ch coalesced gather of x[src],
// fp32 atomic scatter into agg1[dst]; lane0 bumps deg.
__global__ __launch_bounds__(256) void scatter1_kernel(
    const float* __restrict__ x, const int* __restrict__ src,
    const int* __restrict__ dst, float* __restrict__ agg1,
    float* __restrict__ deg, int nE) {
    int lane = threadIdx.x & 31;
    int edge = (blockIdx.x * blockDim.x + threadIdx.x) >> 5;
    if (edge >= nE) return;
    int s = src[edge], d = dst[edge];
    float4 v = *(const float4*)(x + (size_t)s * IN_C + lane * 4);
    float* p = agg1 + (size_t)d * IN_C + lane * 4;
    atomAddF(p + 0, v.x);
    atomAddF(p + 1, v.y);
    atomAddF(p + 2, v.z);
    atomAddF(p + 3, v.w);
    if (lane == 0) atomAddF(deg + d, 1.0f);
}

// ------------------------------------------------------ edge scatter (layer 2)
// gathers f16 hidden activations (half the bytes), scatters fp32 atomics.
__global__ __launch_bounds__(256) void scatter2_kernel(
    const _Float16* __restrict__ h16, const int* __restrict__ src,
    const int* __restrict__ dst, float* __restrict__ agg2, int nE) {
    int lane = threadIdx.x & 31;
    int edge = (blockIdx.x * blockDim.x + threadIdx.x) >> 5;
    if (edge >= nE) return;
    int s = src[edge], d = dst[edge];
    v4h v = *(const v4h*)(h16 + (size_t)s * HID_C + lane * 4);
    float* p = agg2 + (size_t)d * HID_C + lane * 4;
    atomAddF(p + 0, (float)v[0]);
    atomAddF(p + 1, (float)v[1]);
    atomAddF(p + 2, (float)v[2]);
    atomAddF(p + 3, (float)v[3]);
}

// ------------------------------------------------------------- WMMA helpers
// A fragment (16-bit A 16x32): lane holds row m = lane&15; halves i<8 cover
// K = kbase + half*8 + i, halves i>=8 cover K = kbase + 16 + half*8 + (i-8)
// -> two contiguous 16B runs in LDS.
static __device__ __forceinline__ v16h load_a_frag(const _Float16* rowBase,
                                                   int kbase, int half) {
    const _Float16* p = rowBase + kbase + half * 8;
    v8h lo = *(const v8h*)(p);
    v8h hi = *(const v8h*)(p + 16);
    return __builtin_shufflevector(lo, hi, 0, 1, 2, 3, 4, 5, 6, 7, 8, 9, 10,
                                   11, 12, 13, 14, 15);
}

// ------------------------------------------------------------------- layer 1
// h = relu(mean(agg1) @ W1_l^T + x @ W1_r^T + b1) stored as f16.
// 256 threads = 8 waves; wave w owns output cols [16w,16w+16) of a 16-node tile.
__global__ __launch_bounds__(256) void layer1_kernel(
    const float* __restrict__ x, const float* __restrict__ agg1,
    const float* __restrict__ deg, const float* __restrict__ b1,
    const _Float16* __restrict__ w1l_frag, const _Float16* __restrict__ w1r_frag,
    _Float16* __restrict__ h16) {
    __shared__ __align__(32) _Float16 A1[16 * 136];  // mean aggregate, padded
    __shared__ __align__(32) _Float16 A2[16 * 136];  // root features
    const int tile = blockIdx.x;
    const int tid  = threadIdx.x;

    for (int t = tid; t < 16 * 128; t += 256) {
        int row = t >> 7, col = t & 127;
        int node = tile * 16 + row;
        float inv = 1.0f / fmaxf(deg[node], 1.0f);
        A1[row * 136 + col] = (_Float16)(agg1[(size_t)node * IN_C + col] * inv);
        A2[row * 136 + col] = (_Float16)x[(size_t)node * IN_C + col];
    }
    __syncthreads();

    const int w = tid >> 5, lane = tid & 31;
    const int n = lane & 15, half = lane >> 4;
    float bias = b1[w * 16 + n];
    v8f acc;
#pragma unroll
    for (int r = 0; r < 8; ++r) acc[r] = bias;

#pragma unroll
    for (int kk = 0; kk < 4; ++kk) {
        int kbase = kk * 32;
        v16h a1 = load_a_frag(&A1[n * 136], kbase, half);
        v16h a2 = load_a_frag(&A2[n * 136], kbase, half);
        size_t fb = ((size_t)(w * 4 + kk) * 32 + lane) * 16;
        v16h bl = *(const v16h*)(w1l_frag + fb);
        v16h br = *(const v16h*)(w1r_frag + fb);
        acc = __builtin_amdgcn_wmma_f32_16x16x32_f16(false, a1, false, bl,
                                                     (short)0, acc, false, false);
        acc = __builtin_amdgcn_wmma_f32_16x16x32_f16(false, a2, false, br,
                                                     (short)0, acc, false, false);
    }
    // D layout: acc[r] -> row m = r + 8*half, col = w*16 + n. ReLU + f16 store.
#pragma unroll
    for (int r = 0; r < 8; ++r) {
        int node = tile * 16 + r + 8 * half;
        h16[(size_t)node * HID_C + w * 16 + n] = (_Float16)fmaxf(acc[r], 0.0f);
    }
}

// ------------------------------------------------------------------- layer 2
// out = log_softmax(mean(agg2) @ W2_l^T + h @ W2_r^T + b2). 128 thr = 4 waves,
// wave w owns cols [16w,16w+16) of 64; softmax staged through LDS.
__global__ __launch_bounds__(128) void layer2_kernel(
    const _Float16* __restrict__ h16, const float* __restrict__ agg2,
    const float* __restrict__ deg, const float* __restrict__ b2,
    const _Float16* __restrict__ w2l_frag, const _Float16* __restrict__ w2r_frag,
    float* __restrict__ out) {
    __shared__ __align__(32) _Float16 A1[16 * 136];
    __shared__ __align__(32) _Float16 A2[16 * 136];
    __shared__ float outt[16 * 64];
    const int tile = blockIdx.x;
    const int tid  = threadIdx.x;

    for (int t = tid; t < 16 * 128; t += 128) {
        int row = t >> 7, col = t & 127;
        int node = tile * 16 + row;
        float inv = 1.0f / fmaxf(deg[node], 1.0f);
        A1[row * 136 + col] = (_Float16)(agg2[(size_t)node * HID_C + col] * inv);
        A2[row * 136 + col] = h16[(size_t)node * HID_C + col];
    }
    __syncthreads();

    const int w = tid >> 5, lane = tid & 31;
    const int n = lane & 15, half = lane >> 4;
    float bias = b2[w * 16 + n];
    v8f acc;
#pragma unroll
    for (int r = 0; r < 8; ++r) acc[r] = bias;

#pragma unroll
    for (int kk = 0; kk < 4; ++kk) {
        int kbase = kk * 32;
        v16h a1 = load_a_frag(&A1[n * 136], kbase, half);
        v16h a2 = load_a_frag(&A2[n * 136], kbase, half);
        size_t fb = ((size_t)(w * 4 + kk) * 32 + lane) * 16;
        v16h bl = *(const v16h*)(w2l_frag + fb);
        v16h br = *(const v16h*)(w2r_frag + fb);
        acc = __builtin_amdgcn_wmma_f32_16x16x32_f16(false, a1, false, bl,
                                                     (short)0, acc, false, false);
        acc = __builtin_amdgcn_wmma_f32_16x16x32_f16(false, a2, false, br,
                                                     (short)0, acc, false, false);
    }
#pragma unroll
    for (int r = 0; r < 8; ++r)
        outt[(r + 8 * half) * 64 + w * 16 + n] = acc[r];
    __syncthreads();

    if (tid < 16) {
        int node = tile * 16 + tid;
        const float* rowp = &outt[tid * 64];
        float mx = rowp[0];
        for (int c = 1; c < 64; ++c) mx = fmaxf(mx, rowp[c]);
        float s = 0.0f;
        for (int c = 0; c < 64; ++c) s += expf(rowp[c] - mx);
        float lse = mx + logf(s);
        for (int c = 0; c < 64; ++c)
            out[(size_t)node * 64 + c] = rowp[c] - lse;
    }
}

// ---------------------------------------------------------------------- launch
extern "C" void kernel_launch(void* const* d_in, const int* in_sizes, int n_in,
                              void* d_out, int out_size, void* d_ws, size_t ws_size,
                              hipStream_t stream) {
    const float* x   = (const float*)d_in[0];
    const int*   ei  = (const int*)d_in[1];
    const int*   src = ei;
    const int*   dst = ei + N_EDGES;
    const float* W1l = (const float*)d_in[2];
    const float* W1r = (const float*)d_in[3];
    const float* b1  = (const float*)d_in[4];
    const float* W2l = (const float*)d_in[5];
    const float* W2r = (const float*)d_in[6];
    const float* b2  = (const float*)d_in[7];
    float* out = (float*)d_out;

    // workspace layout (256B aligned); deg/agg1/agg2 contiguous so one zero pass
    char* ws = (char*)d_ws;
    size_t off = 0;
    float* deg  = (float*)(ws + off); off += ((size_t)N_NODES * 4 + 255) & ~255ULL;
    float* agg1 = (float*)(ws + off); off += (size_t)N_NODES * IN_C * 4;
    float* agg2 = (float*)(ws + off); off += (size_t)N_NODES * HID_C * 4;
    size_t zero_bytes = off;                       // deg + agg1 + agg2
    _Float16* h16  = (_Float16*)(ws + off); off += (size_t)N_NODES * HID_C * 2;
    _Float16* w1lf = (_Float16*)(ws + off); off += (size_t)HID_C * IN_C * 2;
    _Float16* w1rf = (_Float16*)(ws + off); off += (size_t)HID_C * IN_C * 2;
    _Float16* w2lf = (_Float16*)(ws + off); off += (size_t)OUT_C * HID_C * 2;
    _Float16* w2rf = (_Float16*)(ws + off); off += (size_t)OUT_C * HID_C * 2;
    (void)ws_size; (void)in_sizes; (void)n_in; (void)out_size;

    // 1) zero the atomic accumulators
    long n4 = (long)(zero_bytes / 16);
    zero4_kernel<<<2048, 256, 0, stream>>>((float4*)ws, n4);

    // 2) pack weights into WMMA B-fragment order (f16)
    pack_weights_kernel<<<64, 256, 0, stream>>>(W1l, w1lf, HID_C, IN_C);
    pack_weights_kernel<<<64, 256, 0, stream>>>(W1r, w1rf, HID_C, IN_C);
    pack_weights_kernel<<<32, 256, 0, stream>>>(W2l, w2lf, OUT_C, HID_C);
    pack_weights_kernel<<<32, 256, 0, stream>>>(W2r, w2rf, OUT_C, HID_C);

    // 3) layer-1 aggregation (one wave per edge)
    int scatterBlocks = (N_EDGES * 32 + 255) / 256;   // 100000
    scatter1_kernel<<<scatterBlocks, 256, 0, stream>>>(x, src, dst, agg1, deg,
                                                       N_EDGES);
    // 4) layer-1 GEMM + relu -> f16 hidden
    layer1_kernel<<<N_NODES / 16, 256, 0, stream>>>(x, agg1, deg, b1, w1lf,
                                                    w1rf, h16);
    // 5) layer-2 aggregation on f16 hidden
    scatter2_kernel<<<scatterBlocks, 256, 0, stream>>>(h16, src, dst, agg2,
                                                       N_EDGES);
    // 6) layer-2 GEMM + log_softmax
    layer2_kernel<<<N_NODES / 16, 128, 0, stream>>>(h16, agg2, deg, b2, w2lf,
                                                    w2rf, out);
}
